// TransformerLayer_3513283248692
// MI455X (gfx1250) — matrix operand order repaired and checked by reading, not compiled, not run on hardware
//
#include <hip/hip_runtime.h>

// ---------------------------------------------------------------------------
// Transformer layer (B=2,S=2048,D=1024,H=16,HD=64), bf16 WMMA w/ f32 accum.
// Weights pre-transposed to [N,K] at cast time; GEMM + attention use
// register-staged double-buffered LDS pipelines. Branch-free fast
// sigmoid*tanh from a single v_exp_f32.
// ---------------------------------------------------------------------------

typedef __attribute__((ext_vector_type(16))) __bf16 v16bf;
typedef __attribute__((ext_vector_type(8)))  float  v8f;

#define BS_ROWS 4096   // B*S
#define DM      1024   // D
#define SEQ     2048
#define NH      16
#define HD      64

union FragA { v16bf v; uint4 q[2]; };            // 16 bf16 = 8 VGPRs
union FragC { v8f   v; float f[8]; };            // 16x16 f32 accumulator
union H8    { uint4 q; unsigned short e[8]; };   // 8 bf16 lanes

__device__ __forceinline__ unsigned short f32_to_bf16(float x) {
  unsigned u = __builtin_bit_cast(unsigned, x);
  unsigned r = u + 0x7FFFu + ((u >> 16) & 1u);   // round-to-nearest-even
  return (unsigned short)(r >> 16);
}

__device__ __forceinline__ v8f wmma_bf16(const FragA& a, const FragA& b, v8f c) {
  return __builtin_amdgcn_wmma_f32_16x16x32_bf16(
      false, a.v, false, b.v, (short)0, c, false, false);
}

// branch-free sigmoid(l)*tanh(l): one v_exp_f32 + two v_rcp_f32
__device__ __forceinline__ float sig_tanh(float l) {
  float t  = __builtin_amdgcn_exp2f(-1.44269504f * l);   // e^-l
  float sg = __builtin_amdgcn_rcpf(1.f + t);             // sigmoid(l)
  float t2 = t * t;                                      // e^-2l
  float th = (1.f - t2) * __builtin_amdgcn_rcpf(1.f + t2); // tanh(l)
  return sg * th;
}

// ---------------------------------------------------------------------------
// f32 -> bf16 cast (vectorized x4) : for activations (x)
// ---------------------------------------------------------------------------
__global__ __launch_bounds__(256)
void cast_f32_bf16_k(const float* __restrict__ in, unsigned short* __restrict__ out, int n4) {
  int i = blockIdx.x * 256 + threadIdx.x;
  if (i < n4) {
    float4 f = reinterpret_cast<const float4*>(in)[i];
    ushort4 o;
    o.x = f32_to_bf16(f.x); o.y = f32_to_bf16(f.y);
    o.z = f32_to_bf16(f.z); o.w = f32_to_bf16(f.w);
    reinterpret_cast<ushort4*>(out)[i] = o;
  }
}

// ---------------------------------------------------------------------------
// f32 [K,N] -> bf16 [N,K] tiled transpose cast : for weights (one-time)
// ---------------------------------------------------------------------------
__global__ __launch_bounds__(256)
void cast_transpose_f32_bf16(const float* __restrict__ W, unsigned short* __restrict__ Wt,
                             int K, int N) {
  __shared__ __attribute__((aligned(16))) unsigned short tile[32][40];
  const int tx = threadIdx.x & 31, ty = threadIdx.x >> 5;   // 32 x 8
  const int k0 = blockIdx.y * 32, n0 = blockIdx.x * 32;
#pragma unroll
  for (int i = ty; i < 32; i += 8)     // read coalesced rows of W, store transposed
    tile[tx][i] = f32_to_bf16(W[(size_t)(k0 + i) * N + n0 + tx]);
  __syncthreads();
#pragma unroll
  for (int i = ty; i < 32; i += 8)     // write coalesced rows of Wt
    Wt[(size_t)(n0 + i) * K + k0 + tx] = tile[i][tx];
}

// ---------------------------------------------------------------------------
// C = act(A @ B + bias); out = HAS_RES ? residual + scale*C : C
// A: [M,K] bf16 row-major, Bt: [N,K] bf16 (pre-transposed weight).
// 128x128x32 tile / 8 wave32; register-staged double-buffered LDS.
// All mode flags are compile-time -> branch-free epilogue.
// ---------------------------------------------------------------------------
template <int ACT, bool HAS_RES, bool WRITE_F, bool WRITE_B>
__global__ __launch_bounds__(256)
void gemm_bf16_wmma(const unsigned short* __restrict__ A,
                    const unsigned short* __restrict__ Bt,
                    const float* __restrict__ bias,
                    const float* __restrict__ residual,
                    const float* __restrict__ scale_p,
                    float* __restrict__ outf,
                    unsigned short* __restrict__ outb,
                    int M, int N, int K)
{
  __shared__ __attribute__((aligned(16))) unsigned short As[2][128][40]; // [m][k]
  __shared__ __attribute__((aligned(16))) unsigned short Bs[2][128][40]; // [n][k]

  const int t    = threadIdx.x;
  const int lane = t & 31;
  const int wv   = t >> 5;
  const int r    = lane & 15;
  const int hl   = lane >> 4;
  const int m0 = blockIdx.y * 128, n0 = blockIdx.x * 128;
  const int mbase = (wv & 1) * 64, nbase = (wv >> 1) * 32;

  FragC acc[4][2];
#pragma unroll
  for (int mi = 0; mi < 4; ++mi)
#pragma unroll
    for (int ni = 0; ni < 2; ++ni)
#pragma unroll
      for (int e = 0; e < 8; ++e) acc[mi][ni].f[e] = 0.f;

  // staging: thread -> (row, 16-elem half) for both A and Bt (identical pattern)
  const int row2 = t >> 1, half = t & 1;
  const unsigned short* aSrc = A  + (size_t)(m0 + row2) * K + half * 16;
  const unsigned short* bSrc = Bt + (size_t)(n0 + row2) * K + half * 16;

  uint4 ra0, ra1, rb0, rb1;
  auto load_regs = [&](int kOff) {
    ra0 = *(const uint4*)(aSrc + kOff);
    ra1 = *(const uint4*)(aSrc + kOff + 8);
    rb0 = *(const uint4*)(bSrc + kOff);
    rb1 = *(const uint4*)(bSrc + kOff + 8);
  };
  auto store_lds = [&](int p) {
    *(uint4*)&As[p][row2][half * 16]     = ra0;
    *(uint4*)&As[p][row2][half * 16 + 8] = ra1;
    *(uint4*)&Bs[p][row2][half * 16]     = rb0;
    *(uint4*)&Bs[p][row2][half * 16 + 8] = rb1;
  };

  load_regs(0);
  store_lds(0);

  const int nk = K / 32;
  for (int kt = 0; kt < nk; ++kt) {
    __syncthreads();
    const int p = kt & 1;
    if (kt + 1 < nk) load_regs((kt + 1) * 32);   // prefetch next tile into regs

    FragA a[4], b[2];
#pragma unroll
    for (int mi = 0; mi < 4; ++mi) {   // A frag: lane<16 K{0..7,16..23}; lane>=16 K{8..15,24..31}
      const unsigned short* pp = &As[p][mbase + mi * 16 + r][0];
      a[mi].q[0] = *(const uint4*)(pp + hl * 8);
      a[mi].q[1] = *(const uint4*)(pp + 16 + hl * 8);
    }
#pragma unroll
    for (int ni = 0; ni < 2; ++ni) {   // B frag: lane<16 K{0..7,16..23}; lane>=16 K{8..15,24..31}
      const unsigned short* pp = &Bs[p][nbase + ni * 16 + r][0];
      b[ni].q[0] = *(const uint4*)(pp + hl * 8);
      b[ni].q[1] = *(const uint4*)(pp + 16 + hl * 8);
    }
#pragma unroll
    for (int mi = 0; mi < 4; ++mi)
#pragma unroll
      for (int ni = 0; ni < 2; ++ni)
        acc[mi][ni].v = wmma_bf16(a[mi], b[ni], acc[mi][ni].v);

    if (kt + 1 < nk) store_lds(1 - p);           // fill other buffer
  }

  // ---- branch-free epilogue ----
  float sc = 0.f;
  if constexpr (HAS_RES) sc = *scale_p;
  float bv0 = bias[n0 + nbase + r];
  float bv1 = bias[n0 + nbase + 16 + r];
#pragma unroll
  for (int mi = 0; mi < 4; ++mi) {
#pragma unroll
    for (int ni = 0; ni < 2; ++ni) {
      const float bb = ni ? bv1 : bv0;
      const size_t base =
          (size_t)(m0 + mbase + mi * 16 + hl * 8) * N + (n0 + nbase + ni * 16 + r);
#pragma unroll
      for (int e = 0; e < 8; ++e) {
        float c = acc[mi][ni].f[e] + bb;
        if constexpr (ACT == 1) c = c > 0.f ? c : 0.2f * c;   // leaky_relu(0.2)
        const size_t idx = base + (size_t)e * N;
        float val;
        if constexpr (HAS_RES) val = residual[idx] + sc * c;
        else                   val = c;
        if constexpr (WRITE_F) outf[idx] = val;
        if constexpr (WRITE_B) outb[idx] = f32_to_bf16(val);
      }
    }
  }
}

// ---------------------------------------------------------------------------
// Attention: ctx = (sigmoid(l)*tanh(l)) @ V, l = QK^T/sqrt(D).
// Grid: (S/128 query tiles, B*H). 8 waves, each owns 16 query rows.
// Double-buffered 64-key tiles; batched fragment loads so dscnt waits
// are partial; branch-free activation.
// ---------------------------------------------------------------------------
__global__ __launch_bounds__(256)
void attention_sigtanh(const unsigned short* __restrict__ qm,
                       const unsigned short* __restrict__ km,
                       const unsigned short* __restrict__ vm,
                       unsigned short* __restrict__ om)
{
  __shared__ __attribute__((aligned(16))) unsigned short Ks[2][64][72];    // [key][hd]
  __shared__ __attribute__((aligned(16))) unsigned short Vt[2][64][72];    // [hd][key]
  __shared__ __attribute__((aligned(16))) unsigned short Pt[8][16][72];    // per-wave probs

  const int t    = threadIdx.x;
  const int lane = t & 31;
  const int wv   = t >> 5;
  const int r    = lane & 15;
  const int hl   = lane >> 4;
  const int bh = blockIdx.y;
  const int b = bh >> 4, h = bh & 15;
  const int q0 = blockIdx.x * 128;
  const size_t headoff = (size_t)b * SEQ * DM + (size_t)h * HD;

  // Q fragments for this wave's 16 rows (K-dim = HD = 64 -> 2 fragments)
  FragA aq[2];
  {
    const unsigned short* qrow = qm + headoff + (size_t)(q0 + wv * 16 + r) * DM;
    aq[0].q[0] = *(const uint4*)(qrow + hl * 8);
    aq[0].q[1] = *(const uint4*)(qrow + 16 + hl * 8);
    aq[1].q[0] = *(const uint4*)(qrow + 32 + hl * 8);
    aq[1].q[1] = *(const uint4*)(qrow + 48 + hl * 8);
  }

  FragC cacc[4];
#pragma unroll
  for (int db = 0; db < 4; ++db)
#pragma unroll
    for (int e = 0; e < 8; ++e) cacc[db].f[e] = 0.f;

  // staging maps: K: rows (t>>3) and (t>>3)+32, hd chunk (t&7)*8
  //               V: key pair 2*(t>>3), 2*(t>>3)+1, hd chunk (t&7)*8
  const int srow = t >> 3, sch = t & 7;
  uint4 rk0, rk1;
  H8 rv0, rv1;
  auto load_regs = [&](int kt) {
    const size_t kb = headoff + (size_t)(kt * 64) * DM;
    rk0   = *(const uint4*)(km + kb + (size_t)srow * DM + sch * 8);
    rk1   = *(const uint4*)(km + kb + (size_t)(srow + 32) * DM + sch * 8);
    rv0.q = *(const uint4*)(vm + kb + (size_t)(2 * srow) * DM + sch * 8);
    rv1.q = *(const uint4*)(vm + kb + (size_t)(2 * srow + 1) * DM + sch * 8);
  };
  auto store_lds = [&](int p) {
    *(uint4*)&Ks[p][srow][sch * 8]      = rk0;
    *(uint4*)&Ks[p][srow + 32][sch * 8] = rk1;
#pragma unroll
    for (int j = 0; j < 8; ++j) {       // interleave key pair -> one b32 store
      unsigned u = (unsigned)rv0.e[j] | ((unsigned)rv1.e[j] << 16);
      *(unsigned*)&Vt[p][sch * 8 + j][2 * srow] = u;
    }
  };

  load_regs(0);
  store_lds(0);

  const int nkt = SEQ / 64;
  for (int kt = 0; kt < nkt; ++kt) {
    __syncthreads();
    const int p = kt & 1;
    if (kt + 1 < nkt) load_regs(kt + 1);

    // ---- batched B-fragment loads for all 4 key blocks ----
    FragA bk[4][2];
#pragma unroll
    for (int kb = 0; kb < 4; ++kb) {               // lane = key, K = hd
      const unsigned short* pp = &Ks[p][kb * 16 + r][0];
      bk[kb][0].q[0] = *(const uint4*)(pp + hl * 8);
      bk[kb][0].q[1] = *(const uint4*)(pp + 16 + hl * 8);
      bk[kb][1].q[0] = *(const uint4*)(pp + 32 + hl * 8);
      bk[kb][1].q[1] = *(const uint4*)(pp + 48 + hl * 8);
    }
    // logits (16q x 64k), activation, bf16 probs -> LDS
#pragma unroll
    for (int kb = 0; kb < 4; ++kb) {
      FragC lg;
#pragma unroll
      for (int e = 0; e < 8; ++e) lg.f[e] = 0.f;
      lg.v = wmma_bf16(aq[0], bk[kb][0], lg.v);
      lg.v = wmma_bf16(aq[1], bk[kb][1], lg.v);
#pragma unroll
      for (int e = 0; e < 8; ++e) {
        float pv = sig_tanh(lg.f[e] * 0.03125f);   // 1/sqrt(1024)
        Pt[wv][e + hl * 8][kb * 16 + r] = f32_to_bf16(pv);
      }
    }

    // reload probs in A-fragment layout (same wave: DScnt keeps order)
    FragA ap[2];
    {
      const unsigned short* pp = &Pt[wv][r][0];
      ap[0].q[0] = *(const uint4*)(pp + hl * 8);
      ap[0].q[1] = *(const uint4*)(pp + 16 + hl * 8);
      ap[1].q[0] = *(const uint4*)(pp + 32 + hl * 8);
      ap[1].q[1] = *(const uint4*)(pp + 48 + hl * 8);
    }

    // ---- batched B-fragment loads for all 4 hd blocks ----
    FragA bv[4][2];
#pragma unroll
    for (int db = 0; db < 4; ++db) {               // lane = hd, K = key
      const unsigned short* pp = &Vt[p][db * 16 + r][0];
      bv[db][0].q[0] = *(const uint4*)(pp + hl * 8);
      bv[db][0].q[1] = *(const uint4*)(pp + 16 + hl * 8);
      bv[db][1].q[0] = *(const uint4*)(pp + 32 + hl * 8);
      bv[db][1].q[1] = *(const uint4*)(pp + 48 + hl * 8);
    }
    // ctx += P(16x64) @ V(64x64)
#pragma unroll
    for (int db = 0; db < 4; ++db) {
      cacc[db].v = wmma_bf16(ap[0], bv[db][0], cacc[db].v);
      cacc[db].v = wmma_bf16(ap[1], bv[db][1], cacc[db].v);
    }

    if (kt + 1 < nkt) store_lds(1 - p);
  }

  // write ctx (bf16) into [B*S, D] layout feeding the Wo GEMM
  unsigned short* orow = om + headoff + (size_t)(q0 + wv * 16) * DM;
#pragma unroll
  for (int db = 0; db < 4; ++db)
#pragma unroll
    for (int e = 0; e < 8; ++e)
      orow[(size_t)(e + hl * 8) * DM + db * 16 + r] = f32_to_bf16(cacc[db].f[e]);
}

// ---------------------------------------------------------------------------
// Host orchestration
// ---------------------------------------------------------------------------
extern "C" void kernel_launch(void* const* d_in, const int* in_sizes, int n_in,
                              void* d_out, int out_size, void* d_ws, size_t ws_size,
                              hipStream_t stream)
{
  (void)in_sizes; (void)n_in; (void)out_size; (void)ws_size;
  const float* x     = (const float*)d_in[0];
  const float* W_in  = (const float*)d_in[1];
  const float* b_in  = (const float*)d_in[2];
  const float* Wq    = (const float*)d_in[3];
  const float* bq    = (const float*)d_in[4];
  const float* Wk    = (const float*)d_in[5];
  const float* bk    = (const float*)d_in[6];
  const float* Wv    = (const float*)d_in[7];
  const float* bv    = (const float*)d_in[8];
  const float* Wo    = (const float*)d_in[9];
  const float* bo    = (const float*)d_in[10];
  const float* W1    = (const float*)d_in[11];
  const float* b1    = (const float*)d_in[12];
  const float* W2    = (const float*)d_in[13];
  const float* b2    = (const float*)d_in[14];
  const float* s_att = (const float*)d_in[15];
  const float* s_ff  = (const float*)d_in[16];
  float* out = (float*)d_out;

  const size_t nAct = (size_t)BS_ROWS * DM;
  const size_t nW   = (size_t)DM * DM;

  char* ws = (char*)d_ws;
  size_t off = 0;
  auto grab = [&](size_t bytes) {
    char* p = ws + off;
    off += (bytes + 255) & ~(size_t)255;
    return p;
  };
  unsigned short* xb   = (unsigned short*)grab(nAct * 2);
  unsigned short* wInT = (unsigned short*)grab(nW * 2);
  unsigned short* wQT  = (unsigned short*)grab(nW * 2);
  unsigned short* wKT  = (unsigned short*)grab(nW * 2);
  unsigned short* wVT  = (unsigned short*)grab(nW * 2);
  unsigned short* wOT  = (unsigned short*)grab(nW * 2);
  unsigned short* w1T  = (unsigned short*)grab(nW * 2);
  unsigned short* w2T  = (unsigned short*)grab(nW * 2);
  float*          hF   = (float*)grab(nAct * 4);
  unsigned short* hB   = (unsigned short*)grab(nAct * 2);
  unsigned short* qB   = (unsigned short*)grab(nAct * 2);
  unsigned short* kB   = (unsigned short*)grab(nAct * 2);
  unsigned short* vB   = (unsigned short*)grab(nAct * 2);
  unsigned short* ctxB = (unsigned short*)grab(nAct * 2);
  float*          latF = (float*)grab(nAct * 4);
  unsigned short* latB = (unsigned short*)grab(nAct * 2);
  unsigned short* ffB  = (unsigned short*)grab(nAct * 2);

  // activations: plain cast; weights: fused cast + transpose to [N,K]
  {
    int n4 = (int)(nAct / 4);
    cast_f32_bf16_k<<<dim3((n4 + 255) / 256), dim3(256), 0, stream>>>(x, xb, n4);
  }
  dim3 gt(DM / 32, DM / 32);   // (32, 32)
  cast_transpose_f32_bf16<<<gt, 256, 0, stream>>>(W_in, wInT, DM, DM);
  cast_transpose_f32_bf16<<<gt, 256, 0, stream>>>(Wq,   wQT,  DM, DM);
  cast_transpose_f32_bf16<<<gt, 256, 0, stream>>>(Wk,   wKT,  DM, DM);
  cast_transpose_f32_bf16<<<gt, 256, 0, stream>>>(Wv,   wVT,  DM, DM);
  cast_transpose_f32_bf16<<<gt, 256, 0, stream>>>(Wo,   wOT,  DM, DM);
  cast_transpose_f32_bf16<<<gt, 256, 0, stream>>>(W1,   w1T,  DM, DM);
  cast_transpose_f32_bf16<<<gt, 256, 0, stream>>>(W2,   w2T,  DM, DM);

  dim3 gg(DM / 128, BS_ROWS / 128);  // (8, 32)
  dim3 bb(256);

  // h = x@W_in + b_in   (f32 for residual, bf16 for next GEMMs)
  gemm_bf16_wmma<0, false, true, true><<<gg, bb, 0, stream>>>(
      xb, wInT, b_in, nullptr, nullptr, hF, hB, BS_ROWS, DM, DM);
  // q,k,v projections (bf16 only)
  gemm_bf16_wmma<0, false, false, true><<<gg, bb, 0, stream>>>(
      hB, wQT, bq, nullptr, nullptr, nullptr, qB, BS_ROWS, DM, DM);
  gemm_bf16_wmma<0, false, false, true><<<gg, bb, 0, stream>>>(
      hB, wKT, bk, nullptr, nullptr, nullptr, kB, BS_ROWS, DM, DM);
  gemm_bf16_wmma<0, false, false, true><<<gg, bb, 0, stream>>>(
      hB, wVT, bv, nullptr, nullptr, nullptr, vB, BS_ROWS, DM, DM);

  // attention -> ctx (bf16, [B*S, D])
  dim3 ga(SEQ / 128, 2 * NH);  // (16, 32)
  attention_sigtanh<<<ga, bb, 0, stream>>>(qB, kB, vB, ctxB);

  // latent = h + s_att * (ctx@Wo + bo)
  gemm_bf16_wmma<0, true, true, true><<<gg, bb, 0, stream>>>(
      ctxB, wOT, bo, hF, s_att, latF, latB, BS_ROWS, DM, DM);
  // ff1 = leaky_relu(latent@W1 + b1)
  gemm_bf16_wmma<1, false, false, true><<<gg, bb, 0, stream>>>(
      latB, w1T, b1, nullptr, nullptr, nullptr, ffB, BS_ROWS, DM, DM);
  // out = latent + s_ff * (ff1@W2 + b2)
  gemm_bf16_wmma<0, true, true, false><<<gg, bb, 0, stream>>>(
      ffB, w2T, b2, latF, s_ff, out, nullptr, BS_ROWS, DM, DM);
}
